// Attention_23201413332995
// MI455X (gfx1250) — matrix operand order repaired and checked by reading, not compiled, not run on hardware
//
#include <hip/hip_runtime.h>
#include <hip/hip_bf16.h>
#include <math.h>

// B=16, T=4096, C=1024, D=64. fp32 in/out; internal f16 WMMA w/ f32 accum.

typedef __attribute__((ext_vector_type(16))) _Float16 v16h;
typedef __attribute__((ext_vector_type(8)))  _Float16 v8h;
typedef __attribute__((ext_vector_type(8)))  float    v8f;
typedef __attribute__((ext_vector_type(4)))  float    v4f;

#define BATCH 16
#define TOK   4096
#define DMODEL 1024
#define DHEAD  64

union V16U { v16h v; v8h h[2]; };

__device__ __forceinline__ v8f wmma16(v16h a, v16h b, v8f c) {
  // v_wmma_f32_16x16x32_f16  D = A*B + C
  return __builtin_amdgcn_wmma_f32_16x16x32_f16(
      false, a, false, b, (short)0, c, false, false);
}

// ---------------------------------------------------------------------------
// Kernel 0: pre-swizzle a [C, D] fp32 weight into exact WMMA B-fragment
// layout, f16.
// frag[((kt*4+nt)*32+lane)*16+j] = W[(kt*32 + j + 16*(lane>>4))*64 + nt*16 + (lane&15)]
// ---------------------------------------------------------------------------
__global__ __launch_bounds__(256) void prep_w_kernel(const float* __restrict__ W,
                                                     _Float16* __restrict__ frag) {
  int idx  = blockIdx.x * 256 + threadIdx.x;      // 65536 total
  int j    = idx & 15;
  int lane = (idx >> 4) & 31;
  int nt   = (idx >> 9) & 3;
  int kt   = idx >> 11;
  int kk   = j + 16 * (lane >> 4);
  int n    = nt * 16 + (lane & 15);
  frag[idx] = (_Float16)W[(kt * 32 + kk) * DHEAD + n];
}

// ---------------------------------------------------------------------------
// Kernel 1: fused QKV projection.  x[B*T, C] (fp32) read ONCE, converted to
// f16 in-register; Wq/Wk/Wv consumed as pre-swizzled B-fragments.
// Outputs: qf (f16, pre-scaled by C^-0.5, [B*T][64]), kf (f16, [B*T][64]),
// vT (f16, [B][64][T]) -- transposed so PV B-fragments are contiguous.
// WG = 128 threads = 4 waves; each wave computes 16 rows x 64 cols of q,k,v.
// ---------------------------------------------------------------------------
__global__ __launch_bounds__(128) void qkv_kernel(
    const float* __restrict__ x,
    const _Float16* __restrict__ wqf, const _Float16* __restrict__ wkf,
    const _Float16* __restrict__ wvf,
    _Float16* __restrict__ qf, _Float16* __restrict__ kf,
    _Float16* __restrict__ vT) {
  const int lane = threadIdx.x & 31;
  const int wave = threadIdx.x >> 5;
  const int m    = lane & 15;
  const int g    = lane >> 4;            // half-wave group
  const long row0 = (long)blockIdx.x * 64 + wave * 16;

  v8f accq[4] = {}, acck[4] = {}, accv[4] = {};

  const float* xrow = x + (row0 + m) * (long)DMODEL;

  for (int kt = 0; kt < DMODEL / 32; ++kt) {
    const int K = kt * 32;
    // ---- A fragment: 16x32 f16, per-lane gather (one contiguous 16-elem run) ----
    const float* ap = xrow + K + 16 * g;
    v4f a0 = *(const v4f*)(ap);
    v4f a1 = *(const v4f*)(ap + 4);
    v4f a2 = *(const v4f*)(ap + 8);
    v4f a3 = *(const v4f*)(ap + 12);
    if (kt + 1 < DMODEL / 32) __builtin_prefetch(ap + 32, 0, 1);
    v16h a;
#pragma unroll
    for (int j = 0; j < 4; ++j) {
      a[j]      = (_Float16)a0[j];
      a[j + 4]  = (_Float16)a1[j];
      a[j + 8]  = (_Float16)a2[j];
      a[j + 12] = (_Float16)a3[j];
    }
    // ---- B fragments (pre-swizzled, contiguous 32B per lane) ----
    const long fb = ((long)(kt * 4) * 32 + lane) * 16;
#pragma unroll
    for (int nt = 0; nt < 4; ++nt) {
      const long fo = fb + (long)nt * 32 * 16;
      v16h bq = *(const v16h*)(wqf + fo);
      v16h bk = *(const v16h*)(wkf + fo);
      v16h bv = *(const v16h*)(wvf + fo);
      accq[nt] = wmma16(a, bq, accq[nt]);
      acck[nt] = wmma16(a, bk, acck[nt]);
      accv[nt] = wmma16(a, bv, accv[nt]);
    }
  }

  // ---- epilogue: C/D layout element (M = r + 8g, N = nt*16 + m) ----
  const float qscale = 0.03125f;  // C^-0.5 = 1/32 folded into Q
#pragma unroll
  for (int nt = 0; nt < 4; ++nt) {
    const int col = nt * 16 + m;
#pragma unroll
    for (int r = 0; r < 8; ++r) {
      const long row = row0 + r + 8 * g;
      qf[row * DHEAD + col] = (_Float16)(accq[nt][r] * qscale);
      kf[row * DHEAD + col] = (_Float16)acck[nt][r];
      const long b = row >> 12;          // row / TOK
      const long t = row & (TOK - 1);
      vT[(b * DHEAD + col) * (long)TOK + t] = (_Float16)accv[nt][r];
    }
  }
}

// ---------------------------------------------------------------------------
// Kernel 2: flash attention.  grid = (T/128, B); WG = 256 threads = 8 waves;
// each wave owns 16 query rows, streams keys/values in 64-key chunks (all
// K/V for a batch is L2-resident: 1 MB f16 vs 192 MB L2).
// Row max via 16-lane shuffles; row SUM via a ones-column WMMA (extra
// accumulator rescaled like the output -> zero sum-reduction shuffles).
// P transposed C-layout -> A-layout through LDS (s_wait_dscnt, wave-local).
// ---------------------------------------------------------------------------
__global__ __launch_bounds__(256) void attn_kernel(
    const _Float16* __restrict__ qf, const _Float16* __restrict__ kf,
    const _Float16* __restrict__ vT, float* __restrict__ out) {
  __shared__ __align__(16) _Float16 lds[8][16][72];  // 64 cols + 8 pad
  const int lane = threadIdx.x & 31;
  const int wave = threadIdx.x >> 5;
  const int m    = lane & 15;
  const int g    = lane >> 4;
  const int b    = blockIdx.y;
  const int q0   = blockIdx.x * 128 + wave * 16;  // query row within batch

  // ---- load Q A-fragments (held in regs for whole kernel) ----
  const _Float16* qrow = qf + ((long)b * TOK + q0 + m) * DHEAD;
  V16U qa0, qa1;
  qa0.h[0] = *(const v8h*)(qrow + 16 * g);
  qa0.h[1] = *(const v8h*)(qrow + 16 * g + 8);
  qa1.h[0] = *(const v8h*)(qrow + 32 + 16 * g);
  qa1.h[1] = *(const v8h*)(qrow + 32 + 16 * g + 8);

  // constant all-ones B fragment: row-sum via matrix pipe
  v16h ones;
#pragma unroll
  for (int j = 0; j < 16; ++j) ones[j] = (_Float16)1.0f;

  const _Float16* kbase = kf + (long)b * TOK * DHEAD;
  const _Float16* vbase = vT + (long)b * DHEAD * TOK;

  v8f acc[4] = {};
  v8f accl = {};                 // running softmax denominator (all N equal)
  float mrun[8];
#pragma unroll
  for (int r = 0; r < 8; ++r) mrun[r] = -INFINITY;

  for (int c = 0; c < TOK; c += 64) {
    // ---- S = Q K^T for 64 keys (four 16-key B tiles, two d-halves each) ---
    v8f s[4];
#pragma unroll
    for (int t = 0; t < 4; ++t) {
      const _Float16* kp = kbase + (long)(c + t * 16 + m) * DHEAD + 16 * g;
      v16h kb0 = *(const v16h*)(kp);
      v16h kb1 = *(const v16h*)(kp + 32);
      if (c + 64 < TOK) __builtin_prefetch(kp + 64 * DHEAD, 0, 1);
      v8f z = {};
      z = wmma16(qa0.v, kb0, z);
      s[t] = wmma16(qa1.v, kb1, z);
    }

    // ---- online softmax over this 64-key chunk ----
#pragma unroll
    for (int r = 0; r < 8; ++r) {
      float t0 = fmaxf(fmaxf(s[0][r], s[1][r]), fmaxf(s[2][r], s[3][r]));
#pragma unroll
      for (int off = 1; off < 16; off <<= 1)
        t0 = fmaxf(t0, __shfl_xor(t0, off, 32));    // row max across 16 lanes
      const float mn = fmaxf(mrun[r], t0);
      const float f  = __expf(mrun[r] - mn);
      mrun[r] = mn;
      // rescale running accumulators (incl. denominator column)
      accl[r] *= f;
#pragma unroll
      for (int nt = 0; nt < 4; ++nt) acc[nt][r] *= f;
      // exponentiate + stash P in LDS (row M = r+8g, col = key-in-chunk)
#pragma unroll
      for (int t = 0; t < 4; ++t)
        lds[wave][r + 8 * g][m + 16 * t] = (_Float16)__expf(s[t][r] - mn);
    }
    asm volatile("s_wait_dscnt 0" ::: "memory");

    // ---- reload P as two 16x32 A-fragments (keys 0..31, 32..63) ----
    V16U pa0, pa1;
    pa0.h[0] = *(const v8h*)(&lds[wave][m][16 * g]);
    pa0.h[1] = *(const v8h*)(&lds[wave][m][16 * g + 8]);
    pa1.h[0] = *(const v8h*)(&lds[wave][m][32 + 16 * g]);
    pa1.h[1] = *(const v8h*)(&lds[wave][m][32 + 16 * g + 8]);

    // ---- denominator: l += P * ones  (matrix pipe, no shuffles) ----
    accl = wmma16(pa0.v, ones, accl);
    accl = wmma16(pa1.v, ones, accl);

    // ---- acc += P V  (B frag: lane = d, halfs = key; contiguous in vT) ----
#pragma unroll
    for (int nt = 0; nt < 4; ++nt) {
      const _Float16* vp = vbase + (long)(nt * 16 + m) * TOK + c + 16 * g;
      v16h vb0 = *(const v16h*)(vp);
      v16h vb1 = *(const v16h*)(vp + 32);
      acc[nt] = wmma16(pa0.v, vb0, acc[nt]);
      acc[nt] = wmma16(pa1.v, vb1, acc[nt]);
    }
  }

  // ---- epilogue: out = acc / l  (fp32 [B][T][D]) ----
  float* orow = out + ((long)b * TOK + q0) * DHEAD;
#pragma unroll
  for (int r = 0; r < 8; ++r) {
    const float inv = 1.0f / accl[r];
    const int M = r + 8 * g;
#pragma unroll
    for (int nt = 0; nt < 4; ++nt)
      orow[(long)M * DHEAD + nt * 16 + m] = acc[nt][r] * inv;
  }
}

// ---------------------------------------------------------------------------
extern "C" void kernel_launch(void* const* d_in, const int* in_sizes, int n_in,
                              void* d_out, int out_size, void* d_ws, size_t ws_size,
                              hipStream_t stream) {
  const float* x  = (const float*)d_in[0];
  const float* Wq = (const float*)d_in[1];
  const float* Wk = (const float*)d_in[2];
  const float* Wv = (const float*)d_in[3];
  float* out = (float*)d_out;

  // workspace layout (f16): qf, kf, vT (4M halfs each), 3 W-fragment buffers
  _Float16* ws  = (_Float16*)d_ws;
  const size_t NQ = (size_t)BATCH * TOK * DHEAD;       // 4M halfs
  const size_t NW = (size_t)DMODEL * DHEAD;            // 64K halfs
  _Float16* qf  = ws;
  _Float16* kfp = ws + NQ;
  _Float16* vT  = ws + 2 * NQ;
  _Float16* wqf = ws + 3 * NQ;
  _Float16* wkf = wqf + NW;
  _Float16* wvf = wkf + NW;

  prep_w_kernel<<<NW / 256, 256, 0, stream>>>(Wq, wqf);
  prep_w_kernel<<<NW / 256, 256, 0, stream>>>(Wk, wkf);
  prep_w_kernel<<<NW / 256, 256, 0, stream>>>(Wv, wvf);

  // 65536 rows / 64 rows per WG
  qkv_kernel<<<(BATCH * TOK) / 64, 128, 0, stream>>>(x, wqf, wkf, wvf,
                                                     qf, kfp, vT);

  attn_kernel<<<dim3(TOK / 128, BATCH), 256, 0, stream>>>(qf, kfp, vT, out);
}
